// TheoryAutoregKStepsModel_16621523436002
// MI455X (gfx1250) — 8-bit matrix operand order repaired and checked by reading, compiled, not run on hardware
//
#include <hip/hip_runtime.h>


typedef __attribute__((ext_vector_type(8))) int v8i;

#define TAK_NVARS 128
#define TAK_EMBED 256
#define TAK_L0    1024
#define TAK_STEPS 16
#define TAK_LAMBD 16384.0f
#define TAK_RHO   16384.0f

#define XF_IDX(t, c, ln)  (((((t) * 2)  + (c))  * 32 + (ln)) * 8)
#define X2_IDX(jt, mc, ln) (((((jt) * 16) + (mc)) * 32 + (ln)) * 8)

__global__ __launch_bounds__(256)
void tak_kernel(const float* __restrict__ tok,
                float* __restrict__ out)
{
  __shared__ int xfA[64 * 2 * 32 * 8];
  __shared__ int x2A[8 * 16 * 32 * 8];
  __shared__ float dots[1056];
  __shared__ __align__(16) unsigned char wu8[1088];
  __shared__ __align__(16) unsigned char qm8[128];
  __shared__ unsigned char succ8[TAK_STEPS * TAK_NVARS];
  __shared__ float qsh[TAK_NVARS];
  __shared__ float red[256];

  const int tid  = threadIdx.x;
  const int lane = tid & 31;
  const int wid  = tid >> 5;
  const int b    = blockIdx.x;
  const int nIdx   = lane & 15;
  const int hiHalf = (lane >= 16) ? 1 : 0;
  const size_t tokB = (size_t)b * TAK_L0 * TAK_EMBED;


  for (int id = tid; id < 64 * 2 * 32; id += 256) {
    const int ln = id & 31;
    const int fc = id >> 5;
    const int c  = fc & 1, t = fc >> 1;
    const int row = t * 16 + (ln & 15);
    const int r0  = c * 64 + ((ln >= 16) ? 8 : 0);
    const float* rp = tok + tokB + (size_t)row * TAK_EMBED;
    int frag[8];
#pragma unroll
    for (int run = 0; run < 4; ++run) {
      const int base = r0 + run * 16;
      int w0 = 0, w1 = 0;
#pragma unroll
      for (int e = 0; e < 4; ++e) {
        w0 |= ((rp[base + e]     != 0.0f) ? 1 : 0) << (8 * e);
        w1 |= ((rp[base + 4 + e] != 0.0f) ? 1 : 0) << (8 * e);
      }
      frag[run * 2] = w0; frag[run * 2 + 1] = w1;
    }
    int* dst = &xfA[XF_IDX(t, c, ln)];
#pragma unroll
    for (int k = 0; k < 8; ++k) dst[k] = frag[k];
  }

  for (int id = tid; id < 8 * 16 * 32; id += 256) {
    const int ln = id & 31;
    const int fc = id >> 5;
    const int mc = fc & 15, jt = fc >> 4;
    const int j  = jt * 16 + (ln & 15);
    const int m0 = mc * 64 + ((ln >= 16) ? 8 : 0);
    const float* cp = tok + tokB + TAK_NVARS + j;
    int frag[8];
#pragma unroll
    for (int run = 0; run < 4; ++run) {
      const int base = m0 + run * 16;
      int w0 = 0, w1 = 0;
#pragma unroll
      for (int e = 0; e < 4; ++e) {
        w0 |= ((cp[(size_t)(base + e)     * TAK_EMBED] != 0.0f) ? 1 : 0) << (8 * e);
        w1 |= ((cp[(size_t)(base + 4 + e) * TAK_EMBED] != 0.0f) ? 1 : 0) << (8 * e);
      }
      frag[run * 2] = w0; frag[run * 2 + 1] = w1;
    }
    int* dst = &x2A[X2_IDX(jt, mc, ln)];
#pragma unroll
    for (int k = 0; k < 8; ++k) dst[k] = frag[k];
  }

  for (int m = TAK_L0 + tid; m < 1056; m += 256) dots[m] = 0.0f;
  for (int i = tid; i < TAK_STEPS * TAK_NVARS; i += 256) succ8[i] = 0;
  __syncthreads();

  for (int s = 0; s < TAK_STEPS; ++s) {
    const int L = TAK_L0 + s;

    if (tid < TAK_NVARS) {
      const float qv = (s == 0)
          ? tok[tokB + (size_t)(TAK_L0 - 1) * TAK_EMBED + TAK_NVARS + tid]
          : (float)succ8[(s - 1) * TAK_NVARS + tid];
      qsh[tid] = qv;
      qm8[tid] = (qv == 0.0f) ? 0xFFu : 0x00u;
    }
    __syncthreads();

    const int kq = hiHalf ? 8 : 0;
    v8i bq0, bq1;
    {
      const int2 p0 = *(const int2*)&qm8[kq];
      const int2 p1 = *(const int2*)&qm8[kq + 16];
      const int2 p2 = *(const int2*)&qm8[kq + 32];
      const int2 p3 = *(const int2*)&qm8[kq + 48];
      const int2 q0 = *(const int2*)&qm8[64 + kq];
      const int2 q1 = *(const int2*)&qm8[64 + kq + 16];
      const int2 q2 = *(const int2*)&qm8[64 + kq + 32];
      const int2 q3 = *(const int2*)&qm8[64 + kq + 48];
      bq0 = (v8i){p0.x, p0.y, p1.x, p1.y, p2.x, p2.y, p3.x, p3.y};
      bq1 = (v8i){q0.x, q0.y, q1.x, q1.y, q2.x, q2.y, q3.x, q3.y};
    }
    for (int t = wid; t < 64; t += 8) {
      const v8i a0 = *(const v8i*)&xfA[XF_IDX(t, 0, lane)];
      const v8i a1 = *(const v8i*)&xfA[XF_IDX(t, 1, lane)];
      v8i acc = {0, 0, 0, 0, 0, 0, 0, 0};
      acc = __builtin_amdgcn_wmma_i32_16x16x64_iu8(false, a0, true, bq0, acc, false, false);
      acc = __builtin_amdgcn_wmma_i32_16x16x64_iu8(false, a1, true, bq1, acc, false, false);
      if (nIdx == 0) {
        const int mbase = t * 16 + (hiHalf ? 8 : 0);
#pragma unroll
        for (int v = 0; v < 8; ++v) dots[mbase + v] = (float)acc[v];
      }
    }
    __syncthreads();

    float lmax = -3.4e38f;
    for (int m = tid; m < L; m += 256) lmax = fmaxf(lmax, dots[m]);
    red[tid] = lmax;
    __syncthreads();
    for (int off = 128; off > 0; off >>= 1) {
      if (tid < off) red[tid] = fmaxf(red[tid], red[tid + off]);
      __syncthreads();
    }
    const float dmax = red[0];
    __syncthreads();

    float lsum = 0.0f;
    for (int m = tid; m < 1088; m += 256) {
      unsigned char wb = 0;
      if (m < L) {
        const float wv = expf(TAK_LAMBD * (dots[m] - dmax));
        lsum += wv;
        wb = (wv != 0.0f) ? 1u : 0u;
      }
      wu8[m] = wb;
    }
    red[tid] = lsum;
    __syncthreads();
    for (int off = 128; off > 0; off >>= 1) {
      if (tid < off) red[tid] += red[tid + off];
      __syncthreads();
    }
    const float scale = TAK_RHO / red[0];

    v8i dacc = {0, 0, 0, 0, 0, 0, 0, 0};
    for (int mc = 0; mc < 17; ++mc) {
      const int k0 = mc * 64 + kq;
      const int2 w0v = *(const int2*)&wu8[k0];
      const int2 w1v = *(const int2*)&wu8[k0 + 16];
      const int2 w2v = *(const int2*)&wu8[k0 + 32];
      const int2 w3v = *(const int2*)&wu8[k0 + 48];
      const v8i bw = (v8i){w0v.x, w0v.y, w1v.x, w1v.y, w2v.x, w2v.y, w3v.x, w3v.y};
      v8i a;
      if (mc < 16) {
        a = *(const v8i*)&x2A[X2_IDX(wid, mc, lane)];
      } else {
        const int r0 = hiHalf ? 8 : 0;
        const int jj = wid * 16 + nIdx;
        int w0 = 0, w1 = 0;
#pragma unroll
        for (int e = 0; e < 4; ++e) {
          w0 |= (int)succ8[(r0 + e)     * TAK_NVARS + jj] << (8 * e);
          w1 |= (int)succ8[(r0 + 4 + e) * TAK_NVARS + jj] << (8 * e);
        }
        a = (v8i){w0, w1, 0, 0, 0, 0, 0, 0};
      }
      dacc = __builtin_amdgcn_wmma_i32_16x16x64_iu8(false, a, false, bw, dacc, false, false);
    }

    if (nIdx == 0) {
      const int jb = wid * 16 + (hiHalf ? 8 : 0);
#pragma unroll
      for (int v = 0; v < 8; ++v) {
        const int j = jb + v;
        const float z  = qsh[j] + scale * (float)dacc[v];
        const float lg = tanhf(z - 0.5f);
        out[((size_t)b * TAK_STEPS + s) * TAK_NVARS + j] = lg;
        succ8[s * TAK_NVARS + j] = (lg > 0.0f) ? 1u : 0u;
      }
    }
    __syncthreads();
  }
}

extern "C" void kernel_launch(void* const* d_in, const int* in_sizes, int n_in,
                              void* d_out, int out_size, void* d_ws, size_t ws_size,
                              hipStream_t stream) {
  (void)in_sizes; (void)n_in; (void)out_size; (void)d_ws; (void)ws_size;
  const float* tok = (const float*)d_in[0];
  float* out = (float*)d_out;

  tak_kernel<<<64, 256, 0, stream>>>(tok, out);
}
